// SampleGraphNetwork_34565896798313
// MI455X (gfx1250) — compile-verified, matrix operand order checked by reading, not run on hardware
//
#include <hip/hip_runtime.h>
#include <hip/hip_bf16.h>
#include <cstddef>

// ---------------- problem constants ----------------
constexpr int NB  = 4096;   // batch
constexpr int ND  = 128;    // feature dim (64+16+16+16+16)
constexpr int NH  = 256;    // hidden
constexpr int NT  = 64;     // time-emb dim
constexpr int NK  = 10;     // knn K
constexpr int NMSG = 2 * NH + ND + NT;  // 704
constexpr int NE  = NB * NK;            // 40960 edges
constexpr int RCHUNK = 1024;            // rows per edge-chunk
constexpr int ECHUNK = RCHUNK * NK;     // 10240 edges per chunk
constexpr float FINF = 3.0e38f;

typedef __attribute__((ext_vector_type(16))) _Float16 v16h;
typedef __attribute__((ext_vector_type(8)))  float    v8f;

union Frag16 { v16h h; float4 q[2]; };

// ---------------- build x_flat (f32 + f16) ----------------
__global__ void build_xflat_kernel(const float* __restrict__ xc, const float* __restrict__ d0,
                                   const float* __restrict__ d1, const float* __restrict__ d2,
                                   const float* __restrict__ xo,
                                   float* __restrict__ xf, _Float16* __restrict__ xfh) {
    int i = blockIdx.x;
    int q = threadIdx.x;  // 128
    float v;
    if      (q < 64)  v = xc[i * 64 + q];
    else if (q < 80)  v = d0[i * 16 + (q - 64)];
    else if (q < 96)  v = d1[i * 16 + (q - 80)];
    else if (q < 112) v = d2[i * 16 + (q - 96)];
    else              v = xo[i * 16 + (q - 112)];
    xf[(size_t)i * ND + q] = v;
    xfh[(size_t)i * ND + q] = (_Float16)v;
}

// ---------------- generic f32 -> f16 cast ----------------
__global__ void cast_h_kernel(const float* __restrict__ in, _Float16* __restrict__ out, int n) {
    int t = blockIdx.x * 256 + threadIdx.x;
    if (t < n) out[t] = (_Float16)in[t];
}

// ---------------- f32 W[K][N] -> f16 Wt[N][K] (one-time, tiny) ----------------
__global__ void transpose_cast_kernel(const float* __restrict__ in, _Float16* __restrict__ out,
                                      int K, int N) {
    int t = blockIdx.x * 256 + threadIdx.x;
    if (t < K * N) {
        int k = t / N, n = t - k * N;
        out[(size_t)n * K + k] = (_Float16)in[t];
    }
}

// ---------------- fused L1 cdist + top-K + softmax weights ----------------
// grid = NB/32 blocks, 256 threads. 32 rows per block, candidate tiles of 64.
__global__ __launch_bounds__(256) void knn_kernel(const float* __restrict__ xf,
                                                  int* __restrict__ knn_idx,
                                                  float* __restrict__ knn_w) {
    constexpr int RT = 32, CT = 64;
    __shared__ float4 rowA4[RT][32];       // 16 KB
    __shared__ float4 cand4[CT][33];       // 33.8 KB (padded to dodge bank conflicts)
    const int tid = threadIdx.x;
    const int rbase = blockIdx.x * RT;
    const float4* xf4 = (const float4*)xf;

    for (int i = tid; i < RT * 32; i += 256)
        rowA4[i >> 5][i & 31] = xf4[(size_t)(rbase + (i >> 5)) * 32 + (i & 31)];

    const int r = tid >> 3;     // 0..31  row in tile
    const int sub = tid & 7;    // 0..7   candidate sub-slot
    float bestd[NK]; int besti[NK];
#pragma unroll
    for (int k = 0; k < NK; k++) { bestd[k] = FINF; besti[k] = 0; }

    for (int cb = 0; cb < NB; cb += CT) {
        __syncthreads();
        for (int i = tid; i < CT * 32; i += 256)
            cand4[i >> 5][i & 31] = xf4[(size_t)(cb + (i >> 5)) * 32 + (i & 31)];
        __syncthreads();
        for (int c = sub; c < CT; c += 8) {
            int j = cb + c;
            float d = 0.f;
#pragma unroll 8
            for (int q = 0; q < 32; q++) {
                float4 a = rowA4[r][q], b = cand4[c][q];
                d += fabsf(a.x - b.x) + fabsf(a.y - b.y) + fabsf(a.z - b.z) + fabsf(a.w - b.w);
            }
            if (j == rbase + r) d = FINF;
            if (d < bestd[NK - 1]) {            // insert via one unrolled bubble pass
                bestd[NK - 1] = d; besti[NK - 1] = j;
#pragma unroll
                for (int k = NK - 1; k > 0; k--) {
                    if (bestd[k] < bestd[k - 1]) {
                        float td = bestd[k]; bestd[k] = bestd[k - 1]; bestd[k - 1] = td;
                        int ti = besti[k]; besti[k] = besti[k - 1]; besti[k - 1] = ti;
                    }
                }
            }
        }
    }
    __syncthreads();
    // merge 8 partial lists per row through LDS (reuse cand4 storage)
    float* md = (float*)&cand4[0][0];          // RT*8*NK floats
    int*   mi = (int*)(md + RT * 8 * NK);      // RT*8*NK ints
#pragma unroll
    for (int k = 0; k < NK; k++) {
        md[(r * 8 + sub) * NK + k] = bestd[k];
        mi[(r * 8 + sub) * NK + k] = besti[k];
    }
    __syncthreads();
    if (tid < RT) {
        int row = tid;
        float outd[NK]; int outi[NK];
        for (int k = 0; k < NK; k++) {
            float bm = FINF; int bp = 0;
            for (int pidx = 0; pidx < 8 * NK; pidx++) {
                float dv = md[row * 8 * NK + pidx];
                if (dv < bm) { bm = dv; bp = pidx; }
            }
            outd[k] = bm; outi[k] = mi[row * 8 * NK + bp];
            md[row * 8 * NK + bp] = FINF;
        }
        // softmax over -d (outd ascending -> outd[0] gives the max of -d)
        float m0 = -outd[0], sum = 0.f, w[NK];
#pragma unroll
        for (int k = 0; k < NK; k++) { w[k] = __expf(-outd[k] - m0); sum += w[k]; }
        float inv = 1.f / sum;
#pragma unroll
        for (int k = 0; k < NK; k++) {
            knn_idx[(size_t)(rbase + row) * NK + k] = outi[k];
            knn_w  [(size_t)(rbase + row) * NK + k] = w[k] * inv;
        }
    }
}

// ---------------- WMMA GEMM: out = act(A[MxK] @ Wt[NxK]^T + bias) (+resid) ----------------
// block = 256 (8 waves) computes a 64x64 tile; each wave -> 16 rows x 32 cols (2 accums).
// Double-buffered LDS stages of K=64 -> 4 WMMAs per wave per barrier.
__global__ __launch_bounds__(256) void gemm_kernel(const _Float16* __restrict__ A,
                                                   const _Float16* __restrict__ Wt, // [N][K]
                                                   const float* __restrict__ bias,
                                                   float* __restrict__ outF,
                                                   _Float16* __restrict__ outH,
                                                   const float* __restrict__ resid,
                                                   int M, int N, int K, int gelu_act) {
    __shared__ _Float16 As[2][64][72];   // padded rows (144B stride)
    __shared__ _Float16 Bs[2][64][72];
    const int tid  = threadIdx.x;
    const int wave = tid >> 5;
    const int lane = tid & 31;
    const int wrow = wave >> 1;            // 0..3 (16-row subtile)
    const int wcol = wave & 1;             // 0..1 (32-col subtile)
    const int mbase = blockIdx.y * 64;
    const int nbase = blockIdx.x * 64;
    const int half = lane >> 4;
    const int mr   = lane & 15;
    v8f acc0 = {}; v8f acc1 = {};

    const int lrow = tid >> 3;             // 0..31 (+32 on 2nd pass)
    const int lc8  = (tid & 7) * 8;        // 0..56

    const int nstage = K >> 6;

    // stage loader: 64x64 halves per operand, all b128 traffic
#define LOAD_STAGE(pb, k0)                                                              \
    {                                                                                   \
        _Pragma("unroll")                                                               \
        for (int j = 0; j < 2; j++) {                                                   \
            int row = lrow + j * 32;                                                    \
            *(float4*)&As[pb][row][lc8] =                                               \
                *(const float4*)(A + (size_t)(mbase + row) * K + (k0) + lc8);           \
            *(float4*)&Bs[pb][row][lc8] =                                               \
                *(const float4*)(Wt + (size_t)(nbase + row) * K + (k0) + lc8);          \
        }                                                                               \
    }

    LOAD_STAGE(0, 0);
    __syncthreads();
    for (int st = 0; st < nstage; st++) {
        const int pb = st & 1;
        if (st + 1 < nstage) LOAD_STAGE(pb ^ 1, (st + 1) << 6);
#pragma unroll
        for (int ks = 0; ks < 64; ks += 32) {
            Frag16 a, b0, b1;
            a.q[0]  = *(const float4*)&As[pb][wrow * 16 + mr][ks + half * 8];
            a.q[1]  = *(const float4*)&As[pb][wrow * 16 + mr][ks + 16 + half * 8];
            b0.q[0] = *(const float4*)&Bs[pb][wcol * 32 + mr][ks + half * 8];
            b0.q[1] = *(const float4*)&Bs[pb][wcol * 32 + mr][ks + 16 + half * 8];
            b1.q[0] = *(const float4*)&Bs[pb][wcol * 32 + 16 + mr][ks + half * 8];
            b1.q[1] = *(const float4*)&Bs[pb][wcol * 32 + 16 + mr][ks + 16 + half * 8];
            acc0 = __builtin_amdgcn_wmma_f32_16x16x32_f16(false, a.h, false, b0.h,
                                                          (short)0, acc0, false, false);
            acc1 = __builtin_amdgcn_wmma_f32_16x16x32_f16(false, a.h, false, b1.h,
                                                          (short)0, acc1, false, false);
        }
        __syncthreads();
    }
#undef LOAD_STAGE

    // C layout: lane&15 = col, (lane>>4)*8 + r = row
    const int nloc = lane & 15;
    const int moff = (lane >> 4) * 8;
    const int row0 = mbase + wrow * 16 + moff;
#pragma unroll
    for (int r = 0; r < 8; r++) {
        int row = row0 + r;
#pragma unroll
        for (int t = 0; t < 2; t++) {
            int col = nbase + wcol * 32 + t * 16 + nloc;
            float x = (t == 0 ? acc0[r] : acc1[r]) + bias[col];
            if (gelu_act) x = 0.5f * x * (1.0f + erff(x * 0.70710678118654752f));
            if (resid) x += resid[(size_t)row * N + col];
            if (outF) outF[(size_t)row * N + col] = x;
            if (outH) outH[(size_t)row * N + col] = (_Float16)x;
        }
    }
}

// ---------------- gather edge features -> f16 ----------------
// feat[e] = [ s[src] (256) | s[dst] (256) | xf[src]-xf[dst] (128) | temb[dst] (64) ]
__global__ void gather_feat_kernel(const float* __restrict__ s, const float* __restrict__ xf,
                                   const float* __restrict__ temb, const int* __restrict__ knn_idx,
                                   int ebase, _Float16* __restrict__ feat) {
    int el = blockIdx.x;            // local edge in chunk
    int e  = ebase + el;
    int dst = e / NK;
    int src = knn_idx[e];
    for (int q = threadIdx.x; q < NMSG; q += 256) {
        float v;
        if      (q < NH)          v = s[(size_t)src * NH + q];
        else if (q < 2 * NH)      v = s[(size_t)dst * NH + (q - NH)];
        else if (q < 2 * NH + ND) v = xf[(size_t)src * ND + (q - 2 * NH)] -
                                      xf[(size_t)dst * ND + (q - 2 * NH)];
        else                      v = temb[(size_t)dst * NT + (q - 2 * NH - ND)];
        feat[(size_t)el * NMSG + q] = (_Float16)v;
    }
}

// ---------------- weighted segment-sum over contiguous groups of NK ----------------
__global__ void agg_kernel(const float* __restrict__ msg, const float* __restrict__ ew,
                           int rbase, float* __restrict__ agg) {
    int il = blockIdx.x;            // local row in chunk
    int i  = rbase + il;
    int h  = threadIdx.x;           // 256
    float a = 0.f;
#pragma unroll
    for (int k = 0; k < NK; k++)
        a += ew[(size_t)i * NK + k] * msg[(size_t)(il * NK + k) * NH + h];
    agg[(size_t)i * NH + h] = a;
}

// ---------------- concat [s | agg] -> f16 ----------------
__global__ void concat2_kernel(const float* __restrict__ s, const float* __restrict__ agg,
                               _Float16* __restrict__ cat2) {
    int i = blockIdx.x;
    for (int q = threadIdx.x; q < 2 * NH; q += 256) {
        float v = (q < NH) ? s[(size_t)i * NH + q] : agg[(size_t)i * NH + (q - NH)];
        cat2[(size_t)i * 2 * NH + q] = (_Float16)v;
    }
}

// ---------------- mean-center the categorical blocks ----------------
__global__ void center_kernel(const float* __restrict__ v, float* __restrict__ out) {
    int i = blockIdx.x;
    int q = threadIdx.x;            // 128
    __shared__ float buf[128];
    __shared__ float mean[3];
    float val = v[(size_t)i * ND + q];
    buf[q] = val;
    __syncthreads();
    if (q < 3) {
        float s = 0.f;
        for (int k = 0; k < 16; k++) s += buf[64 + q * 16 + k];
        mean[q] = s * (1.0f / 16.0f);
    }
    __syncthreads();
    float o = val;
    if (q >= 64 && q < 112) o = val - mean[(q - 64) >> 4];
    out[(size_t)i * ND + q] = o;
}

// ---------------- host ----------------
extern "C" void kernel_launch(void* const* d_in, const int* in_sizes, int n_in,
                              void* d_out, int out_size, void* d_ws, size_t ws_size,
                              hipStream_t stream) {
    (void)in_sizes; (void)n_in; (void)out_size; (void)ws_size;
    const float* x_c  = (const float*)d_in[0];
    const float* x_d0 = (const float*)d_in[1];
    const float* x_d1 = (const float*)d_in[2];
    const float* x_d2 = (const float*)d_in[3];
    const float* x_o  = (const float*)d_in[4];
    const float* temb = (const float*)d_in[5];
    const float* W_in = (const float*)d_in[6];
    const float* b_in = (const float*)d_in[7];
    const float* Wm1  = (const float*)d_in[8];
    const float* bm1  = (const float*)d_in[9];
    const float* Wm2  = (const float*)d_in[10];
    const float* bm2  = (const float*)d_in[11];
    const float* Wu1  = (const float*)d_in[12];
    const float* bu1  = (const float*)d_in[13];
    const float* Wu2  = (const float*)d_in[14];
    const float* bu2  = (const float*)d_in[15];
    const float* Wout = (const float*)d_in[16];
    const float* bout = (const float*)d_in[17];
    float* out = (float*)d_out;

    char* p = (char*)d_ws;
    auto alloc = [&](size_t bytes) -> char* {
        char* r = p; p += (bytes + 255) & ~(size_t)255; return r;
    };
    float*    xf    = (float*)   alloc((size_t)NB * ND * 4);
    _Float16* xfh   = (_Float16*)alloc((size_t)NB * ND * 2);
    _Float16* winT  = (_Float16*)alloc((size_t)ND * NH * 2);          // [NH][ND]
    _Float16* wm1T  = (_Float16*)alloc((size_t)2 * NMSG * NH * 2);    // per-layer [NH][NMSG]
    _Float16* wm2T  = (_Float16*)alloc((size_t)2 * NH * NH * 2);
    _Float16* wu1T  = (_Float16*)alloc((size_t)2 * 2 * NH * NH * 2);
    _Float16* wu2T  = (_Float16*)alloc((size_t)2 * NH * NH * 2);
    _Float16* woutT = (_Float16*)alloc((size_t)NH * ND * 2);          // [ND][NH]
    int*      kidx  = (int*)     alloc((size_t)NE * 4);
    float*    ew    = (float*)   alloc((size_t)NE * 4);
    float*    s     = (float*)   alloc((size_t)NB * NH * 4);
    _Float16* feat  = (_Float16*)alloc((size_t)ECHUNK * NMSG * 2);
    _Float16* mh1   = (_Float16*)alloc((size_t)ECHUNK * NH * 2);
    float*    msg   = (float*)   alloc((size_t)ECHUNK * NH * 4);
    float*    agg   = (float*)   alloc((size_t)NB * NH * 4);
    _Float16* cat2  = (_Float16*)alloc((size_t)NB * 2 * NH * 2);
    _Float16* sh    = (_Float16*)alloc((size_t)NB * NH * 2);
    float*    vbuf  = (float*)   alloc((size_t)NB * ND * 4);

    auto castT = [&](const float* src, _Float16* dst, int K, int N) {
        int n = K * N;
        transpose_cast_kernel<<<(n + 255) / 256, 256, 0, stream>>>(src, dst, K, N);
    };
    auto gemm = [&](const _Float16* A, const _Float16* Wt, const float* bias,
                    float* oF, _Float16* oH, const float* resid,
                    int M, int N, int K, int act) {
        dim3 g(N / 64, M / 64);
        gemm_kernel<<<g, 256, 0, stream>>>(A, Wt, bias, oF, oH, resid, M, N, K, act);
    };

    // 1) inputs + weight conversion (pre-transposed to [N][K] f16)
    build_xflat_kernel<<<NB, 128, 0, stream>>>(x_c, x_d0, x_d1, x_d2, x_o, xf, xfh);
    castT(W_in, winT, ND, NH);
    for (int l = 0; l < 2; l++) {
        castT(Wm1 + (size_t)l * NMSG * NH,   wm1T + (size_t)l * NMSG * NH,   NMSG,   NH);
        castT(Wm2 + (size_t)l * NH * NH,     wm2T + (size_t)l * NH * NH,     NH,     NH);
        castT(Wu1 + (size_t)l * 2 * NH * NH, wu1T + (size_t)l * 2 * NH * NH, 2 * NH, NH);
        castT(Wu2 + (size_t)l * NH * NH,     wu2T + (size_t)l * NH * NH,     NH,     NH);
    }
    castT(Wout, woutT, NH, ND);

    // 2) KNN graph (fused L1 cdist + top-K + softmax weights)
    knn_kernel<<<NB / 32, 256, 0, stream>>>(xf, kidx, ew);

    // 3) s = x_flat @ W_in + b_in
    gemm(xfh, winT, b_in, s, nullptr, nullptr, NB, NH, ND, 0);

    // 4) message-passing layers
    for (int l = 0; l < 2; l++) {
        const _Float16* wm1l = wm1T + (size_t)l * NMSG * NH;
        const _Float16* wm2l = wm2T + (size_t)l * NH * NH;
        const _Float16* wu1l = wu1T + (size_t)l * 2 * NH * NH;
        const _Float16* wu2l = wu2T + (size_t)l * NH * NH;
        const float* bm1l = bm1 + l * NH;
        const float* bm2l = bm2 + l * NH;
        const float* bu1l = bu1 + l * NH;
        const float* bu2l = bu2 + l * NH;
        for (int c = 0; c < NB / RCHUNK; c++) {
            gather_feat_kernel<<<ECHUNK, 256, 0, stream>>>(s, xf, temb, kidx,
                                                           c * ECHUNK, feat);
            gemm(feat, wm1l, bm1l, nullptr, mh1, nullptr, ECHUNK, NH, NMSG, 1);
            gemm(mh1,  wm2l, bm2l, msg, nullptr, nullptr, ECHUNK, NH, NH, 0);
            agg_kernel<<<RCHUNK, NH, 0, stream>>>(msg, ew, c * RCHUNK, agg);
        }
        concat2_kernel<<<NB, 256, 0, stream>>>(s, agg, cat2);
        gemm(cat2, wu1l, bu1l, nullptr, sh /*reuse as u1h*/, nullptr, NB, NH, 2 * NH, 1);
        gemm(sh,   wu2l, bu2l, s, nullptr, s /*residual*/, NB, NH, NH, 0);
    }

    // 5) output projection + mean-centering
    cast_h_kernel<<<(NB * NH + 255) / 256, 256, 0, stream>>>(s, sh, NB * NH);
    gemm(sh, woutT, bout, vbuf, nullptr, nullptr, NB, ND, NH, 0);
    center_kernel<<<NB, 128, 0, stream>>>(vbuf, out);
}
